// EncoderDecoder_12008728560047
// MI455X (gfx1250) — matrix operand order repaired and checked by reading, not compiled, not run on hardware
//
#include <hip/hip_runtime.h>

// ---------------------------------------------------------------------------
// Problem dims
// ---------------------------------------------------------------------------
#define BB    32
#define TT    256
#define EMB_D 256
#define CV_D  256
#define K2_D  512
#define VV    32000

typedef __attribute__((ext_vector_type(16))) __bf16 v16bf;
typedef __attribute__((ext_vector_type(8)))  float  v8f;

__device__ __forceinline__ v8f v8f_zero() {
    v8f z = {0.f, 0.f, 0.f, 0.f, 0.f, 0.f, 0.f, 0.f};
    return z;
}

__device__ __forceinline__ v8f wmma_bf16(v16bf a, v16bf b, v8f c) {
    return __builtin_amdgcn_wmma_f32_16x16x32_bf16(
        false, a, false, b, (short)0, c, false, false);
}

// A-fragment layout for 16-bit 16x32 tiles, packed as [Mtile][Ktile][lane][16].
// Per ISA: lanes 0-15 hold M=lane, elements 0-7 -> K 0..7, 8-15 -> K 16..23;
//          lanes 16-31 hold M=lane-16, elements 0-7 -> K 8..15, 8-15 -> K 24..31.
__device__ __forceinline__ size_t a_frag_index(int M, int K, int Ktot) {
    int mt = M >> 4, kt = K >> 5, k = K & 31;
    int lane = (M & 15) + (((k >> 3) & 1) << 4);
    int e = (k & 7) + ((k & 16) ? 8 : 0);
    return (((size_t)(mt * (Ktot >> 5) + kt)) * 32 + lane) * 16 + e;
}

// ---------------------------------------------------------------------------
// Convert fp32 [K,N] weight into bf16 B-fragment layout [Ntile][Ktile][lane][16]
// (lane = (n&15) + 16*((k>>3)&1), e = (k&7) + 8*(k>=16 within 32-chunk)) so the GEMM inner loop
// is one contiguous 32B load per lane.  Native __bf16 cast -> HW cvt op.
// ---------------------------------------------------------------------------
__global__ void convert_b_frag(const float* __restrict__ src,
                               __bf16* __restrict__ dst, int K, int N) {
    int idx = blockIdx.x * 256 + threadIdx.x;
    if (idx >= K * N) return;
    int k = idx / N, n = idx % N;
    int nt = n >> 4, kt = k >> 5;
    int lane = (n & 15) + (((k >> 3) & 1) << 4);
    int e = (k & 7) + ((k & 16) ? 8 : 0);
    size_t o = (((size_t)nt * (K >> 5) + kt) * 32 + lane) * 16 + e;
    dst[o] = (__bf16)src[idx];
}

// ---------------------------------------------------------------------------
// Embedding gather + encoder input projection (both directions).
// M = t*32 + b (8192 rows), K = EMB, N = CV.  Output proj[T][B][CV] fp32.
// ---------------------------------------------------------------------------
__global__ __launch_bounds__(256) void embed_proj(
    const int* __restrict__ tokens, const float* __restrict__ emb,
    const __bf16* __restrict__ wF, const __bf16* __restrict__ wB,
    float* __restrict__ projF, float* __restrict__ projB) {
    int wave = threadIdx.x >> 5, lane = threadIdx.x & 31;
    int dir = blockIdx.z;
    const __bf16* wf = dir ? wB : wF;
    float* proj = dir ? projB : projF;

    int mt = blockIdx.x * 8 + wave;        // 0..511
    int ntbase = blockIdx.y * 8;           // 0 or 8

    int M = mt * 16 + (lane & 15);
    int b = M & 31, t = M >> 5;
    const float* row = emb + (size_t)tokens[b * TT + t] * EMB_D;

    v8f acc[8];
#pragma unroll
    for (int i = 0; i < 8; ++i) acc[i] = v8f_zero();

#pragma unroll
    for (int kt = 0; kt < 8; ++kt) {
        int kb = kt * 32 + ((lane & 16) ? 8 : 0);
        v16bf af;
#pragma unroll
        for (int e = 0; e < 8; ++e) {
            af[e]     = (__bf16)row[kb + e];
            af[8 + e] = (__bf16)row[kb + 16 + e];
        }
#pragma unroll
        for (int nt2 = 0; nt2 < 8; ++nt2) {
            const v16bf bfr = *(const v16bf*)(wf +
                (((size_t)(ntbase + nt2) * 8 + kt) * 32 + lane) * 16);
            acc[nt2] = wmma_bf16(af, bfr, acc[nt2]);
        }
    }

    int mloc = (lane & 16) ? 8 : 0;
#pragma unroll
    for (int nt2 = 0; nt2 < 8; ++nt2) {
        int n = (ntbase + nt2) * 16 + (lane & 15);
#pragma unroll
        for (int r = 0; r < 8; ++r) {
            int m = mt * 16 + r + mloc;
            proj[(size_t)m * CV_D + n] = acc[nt2][r];
        }
    }
}

// ---------------------------------------------------------------------------
// Encoder scan: h = tanh(proj_t + h@wh + b).  One workgroup per direction,
// wh resident in VGPRs as bf16 fragments, h in LDS in A-fragment layout.
// Writes final state into ctx[:, dir*256 : dir*256+256].
// ---------------------------------------------------------------------------
__global__ __launch_bounds__(256) void enc_scan(
    const float* __restrict__ projF, const float* __restrict__ projB,
    const __bf16* __restrict__ whF, const __bf16* __restrict__ whB,
    const float* __restrict__ biasF, const float* __restrict__ biasB,
    float* __restrict__ ctx) {
    __shared__ __attribute__((aligned(32))) __bf16 hfrag[2 * 8 * 32 * 16];

    int wave = threadIdx.x >> 5, lane = threadIdx.x & 31;
    int dir = blockIdx.x;
    const float* proj = dir ? projB : projF;
    const __bf16* wh = dir ? whB : whF;
    const float* bias = dir ? biasB : biasF;

    for (int i = threadIdx.x; i < 2 * 8 * 32 * 16; i += 256)
        hfrag[i] = (__bf16)0.0f;

    // Preload this wave's 2 N-tiles of wh (2 x 8 fragments = 128 VGPRs).
    v16bf Bf[2][8];
#pragma unroll
    for (int nt2 = 0; nt2 < 2; ++nt2)
#pragma unroll
        for (int kt = 0; kt < 8; ++kt)
            Bf[nt2][kt] = *(const v16bf*)(wh +
                (((size_t)(wave * 2 + nt2) * 8 + kt) * 32 + lane) * 16);

    __syncthreads();

    for (int t = 0; t < TT; ++t) {
        int tp = dir ? (TT - 1 - t) : t;
        v8f acc[2][2];
#pragma unroll
        for (int mt = 0; mt < 2; ++mt) { acc[mt][0] = v8f_zero(); acc[mt][1] = v8f_zero(); }

#pragma unroll
        for (int mt = 0; mt < 2; ++mt)
#pragma unroll
            for (int kt = 0; kt < 8; ++kt) {
                v16bf a = *(const v16bf*)&hfrag[((mt * 8 + kt) * 32 + lane) * 16];
                acc[mt][0] = wmma_bf16(a, Bf[0][kt], acc[mt][0]);
                acc[mt][1] = wmma_bf16(a, Bf[1][kt], acc[mt][1]);
            }
        __syncthreads();

        int mloc = (lane & 16) ? 8 : 0;
#pragma unroll
        for (int mt = 0; mt < 2; ++mt)
#pragma unroll
            for (int nt2 = 0; nt2 < 2; ++nt2) {
                int n = (wave * 2 + nt2) * 16 + (lane & 15);
#pragma unroll
                for (int r = 0; r < 8; ++r) {
                    int m = mt * 16 + r + mloc;
                    float v = tanhf(acc[mt][nt2][r] +
                                    proj[(size_t)tp * (BB * CV_D) + m * CV_D + n] +
                                    bias[n]);
                    hfrag[a_frag_index(m, n, CV_D)] = (__bf16)v;
                    if (t == TT - 1) ctx[m * (2 * CV_D) + dir * CV_D + n] = v;
                }
            }
        __syncthreads();
    }
}

// ---------------------------------------------------------------------------
// Decoder constant input projection: p = ctx @ w + b   ([32,512]x[512,256]).
// Tiny; plain fp32 dot products.
// ---------------------------------------------------------------------------
__global__ void dec_proj(const float* __restrict__ ctx,
                         const float* __restrict__ dfwi, const float* __restrict__ dfb,
                         const float* __restrict__ dbwi, const float* __restrict__ dbb,
                         float* __restrict__ pf, float* __restrict__ pb) {
    int idx = blockIdx.x * 256 + threadIdx.x;   // 0..16383
    int dir = idx >> 13;
    int id = idx & 8191;
    int m = id >> 8, n = id & 255;
    const float* wi = dir ? dbwi : dfwi;
    float acc = dir ? dbb[n] : dfb[n];
    for (int k = 0; k < K2_D; ++k)
        acc += ctx[m * K2_D + k] * wi[k * CV_D + n];
    (dir ? pb : pf)[m * CV_D + n] = acc;
}

// ---------------------------------------------------------------------------
// Decoder scan: h = tanh(p + h@wh). Streams h directly into the global bf16
// A-matrix for the Dense GEMM (fragment layout, Ktot=512).
//   fwd: row M = b*256 + t,        K = n
//   bwd: row M = b*256 + (255-t),  K = 256 + n
// ---------------------------------------------------------------------------
__global__ __launch_bounds__(256) void dec_scan(
    const float* __restrict__ pf, const float* __restrict__ pb,
    const __bf16* __restrict__ whF, const __bf16* __restrict__ whB,
    __bf16* __restrict__ decA) {
    __shared__ __attribute__((aligned(32))) __bf16 hfrag[2 * 8 * 32 * 16];

    int wave = threadIdx.x >> 5, lane = threadIdx.x & 31;
    int dir = blockIdx.x;
    const float* pd = dir ? pb : pf;
    const __bf16* wh = dir ? whB : whF;

    for (int i = threadIdx.x; i < 2 * 8 * 32 * 16; i += 256)
        hfrag[i] = (__bf16)0.0f;

    v16bf Bf[2][8];
#pragma unroll
    for (int nt2 = 0; nt2 < 2; ++nt2)
#pragma unroll
        for (int kt = 0; kt < 8; ++kt)
            Bf[nt2][kt] = *(const v16bf*)(wh +
                (((size_t)(wave * 2 + nt2) * 8 + kt) * 32 + lane) * 16);

    __syncthreads();

    for (int t = 0; t < TT; ++t) {
        v8f acc[2][2];
#pragma unroll
        for (int mt = 0; mt < 2; ++mt) { acc[mt][0] = v8f_zero(); acc[mt][1] = v8f_zero(); }

#pragma unroll
        for (int mt = 0; mt < 2; ++mt)
#pragma unroll
            for (int kt = 0; kt < 8; ++kt) {
                v16bf a = *(const v16bf*)&hfrag[((mt * 8 + kt) * 32 + lane) * 16];
                acc[mt][0] = wmma_bf16(a, Bf[0][kt], acc[mt][0]);
                acc[mt][1] = wmma_bf16(a, Bf[1][kt], acc[mt][1]);
            }
        __syncthreads();

        int tout = dir ? (TT - 1 - t) : t;
        int mloc = (lane & 16) ? 8 : 0;
#pragma unroll
        for (int mt = 0; mt < 2; ++mt)
#pragma unroll
            for (int nt2 = 0; nt2 < 2; ++nt2) {
                int n = (wave * 2 + nt2) * 16 + (lane & 15);
#pragma unroll
                for (int r = 0; r < 8; ++r) {
                    int m = mt * 16 + r + mloc;          // batch row 0..31
                    float v = tanhf(acc[mt][nt2][r] + pd[m * CV_D + n]);
                    __bf16 hv = (__bf16)v;
                    hfrag[a_frag_index(m, n, CV_D)] = hv;
                    size_t Mout = (size_t)m * TT + tout;
                    decA[a_frag_index((int)Mout, dir * CV_D + n, K2_D)] = hv;
                }
            }
        __syncthreads();
    }
}

// ---------------------------------------------------------------------------
// Dense: logits = dec_out(bf16 frag) @ dense_w(bf16 frag) + b.  268 GFLOP.
// Workgroup = 1 Mtile x 256 columns (8 waves x 2 Ntiles). Pure b128+WMMA loop
// with L2->near prefetch of upcoming W fragments.
// ---------------------------------------------------------------------------
__global__ __launch_bounds__(256) void dense_kernel(
    const __bf16* __restrict__ Afrag,
    const __bf16* __restrict__ Wfrag,
    const float* __restrict__ bias, float* __restrict__ out) {
    int wave = threadIdx.x >> 5, lane = threadIdx.x & 31;
    int mt = blockIdx.x;                       // 0..511
    int ntbase = blockIdx.y * 16 + wave * 2;   // 0..1998

    const __bf16* Ab = Afrag + (size_t)mt * 16 * 32 * 16;
    v8f acc0 = v8f_zero(), acc1 = v8f_zero();

#pragma unroll
    for (int kt = 0; kt < 16; ++kt) {
        if (kt + 4 < 16) {
            __builtin_prefetch(Wfrag + (((size_t)ntbase * 16 + kt + 4) * 32 + lane) * 16, 0, 0);
            __builtin_prefetch(Wfrag + (((size_t)(ntbase + 1) * 16 + kt + 4) * 32 + lane) * 16, 0, 0);
        }
        v16bf a  = *(const v16bf*)(Ab + ((size_t)kt * 32 + lane) * 16);
        v16bf b0 = *(const v16bf*)(Wfrag + (((size_t)ntbase * 16 + kt) * 32 + lane) * 16);
        v16bf b1 = *(const v16bf*)(Wfrag + (((size_t)(ntbase + 1) * 16 + kt) * 32 + lane) * 16);
        acc0 = wmma_bf16(a, b0, acc0);
        acc1 = wmma_bf16(a, b1, acc1);
    }

    int mloc = (lane & 16) ? 8 : 0;
    int n0 = ntbase * 16 + (lane & 15);
    float bi0 = bias[n0], bi1 = bias[n0 + 16];
#pragma unroll
    for (int r = 0; r < 8; ++r) {
        int M = mt * 16 + r + mloc;
        out[(size_t)M * VV + n0]      = acc0[r] + bi0;
        out[(size_t)M * VV + n0 + 16] = acc1[r] + bi1;
    }
}

// ---------------------------------------------------------------------------
// Row softmax over 32000, in place. Pass 1: online (max,sum). Pass 2: scale.
// ---------------------------------------------------------------------------
__global__ __launch_bounds__(256) void softmax_kernel(float* __restrict__ out) {
    __shared__ float sm[256], ss[256];
    float* p = out + (size_t)blockIdx.x * VV;
    float m = -3.4e38f, s = 0.f;
    for (int i = threadIdx.x; i < VV; i += 256) {
        float x = p[i];
        float mn = fmaxf(m, x);
        s = s * __expf(m - mn) + __expf(x - mn);
        m = mn;
    }
    sm[threadIdx.x] = m; ss[threadIdx.x] = s;
    __syncthreads();
    for (int off = 128; off > 0; off >>= 1) {
        if (threadIdx.x < (unsigned)off) {
            float m2 = sm[threadIdx.x + off], s2 = ss[threadIdx.x + off];
            float mn = fmaxf(sm[threadIdx.x], m2);
            ss[threadIdx.x] = ss[threadIdx.x] * __expf(sm[threadIdx.x] - mn)
                            + s2 * __expf(m2 - mn);
            sm[threadIdx.x] = mn;
        }
        __syncthreads();
    }
    float M = sm[0], inv = 1.0f / ss[0];
    for (int i = threadIdx.x; i < VV; i += 256)
        p[i] = __expf(p[i] - M) * inv;
}

// ---------------------------------------------------------------------------
// Workspace layout (bytes)
// ---------------------------------------------------------------------------
#define OFF_PROJF   ((size_t)0)                       // 8 MB  [T][B][CV] fp32
#define OFF_PROJB   (OFF_PROJF + (size_t)8388608)     // 8 MB
#define OFF_CTX     (OFF_PROJB + (size_t)8388608)     // 64 KB [32][512] fp32
#define OFF_PDF     (OFF_CTX + (size_t)65536)         // 32 KB
#define OFF_PDB     (OFF_PDF + (size_t)32768)         // 32 KB
#define OFF_DECA    (OFF_PDB + (size_t)32768)         // 8 MB  bf16 frag [8192,512]
#define OFF_WDENSE  (OFF_DECA + (size_t)8388608)      // 31.25 MB bf16 frag
#define OFF_FEFWI   (OFF_WDENSE + (size_t)32768000)   // 6 x 128 KB weight frags
#define OFF_FEBWI   (OFF_FEFWI + (size_t)131072)
#define OFF_FEFWH   (OFF_FEBWI + (size_t)131072)
#define OFF_FEBWH   (OFF_FEFWH + (size_t)131072)
#define OFF_FDFWH   (OFF_FEBWH + (size_t)131072)
#define OFF_FDBWH   (OFF_FDFWH + (size_t)131072)

extern "C" void kernel_launch(void* const* d_in, const int* in_sizes, int n_in,
                              void* d_out, int out_size, void* d_ws, size_t ws_size,
                              hipStream_t stream) {
    (void)in_sizes; (void)n_in; (void)out_size; (void)ws_size;
    const int*   tokens  = (const int*)d_in[0];
    const float* emb     = (const float*)d_in[1];
    const float* efwi    = (const float*)d_in[2];
    const float* efwh    = (const float*)d_in[3];
    const float* efb     = (const float*)d_in[4];
    const float* ebwi    = (const float*)d_in[5];
    const float* ebwh    = (const float*)d_in[6];
    const float* ebb     = (const float*)d_in[7];
    const float* dfwi    = (const float*)d_in[8];
    const float* dfwh    = (const float*)d_in[9];
    const float* dfb     = (const float*)d_in[10];
    const float* dbwi    = (const float*)d_in[11];
    const float* dbwh    = (const float*)d_in[12];
    const float* dbb     = (const float*)d_in[13];
    const float* dense_w = (const float*)d_in[14];
    const float* dense_b = (const float*)d_in[15];
    float* out = (float*)d_out;
    char*  ws  = (char*)d_ws;

    float* projF = (float*)(ws + OFF_PROJF);
    float* projB = (float*)(ws + OFF_PROJB);
    float* ctx   = (float*)(ws + OFF_CTX);
    float* pdf   = (float*)(ws + OFF_PDF);
    float* pdb   = (float*)(ws + OFF_PDB);
    __bf16* decA    = (__bf16*)(ws + OFF_DECA);
    __bf16* wdense  = (__bf16*)(ws + OFF_WDENSE);
    __bf16* f_efwi  = (__bf16*)(ws + OFF_FEFWI);
    __bf16* f_ebwi  = (__bf16*)(ws + OFF_FEBWI);
    __bf16* f_efwh  = (__bf16*)(ws + OFF_FEFWH);
    __bf16* f_ebwh  = (__bf16*)(ws + OFF_FEBWH);
    __bf16* f_dfwh  = (__bf16*)(ws + OFF_FDFWH);
    __bf16* f_dbwh  = (__bf16*)(ws + OFF_FDBWH);

    // 1) Weight conversions to bf16 fragment layout.
    convert_b_frag<<<256, 256, 0, stream>>>(efwi, f_efwi, 256, 256);
    convert_b_frag<<<256, 256, 0, stream>>>(ebwi, f_ebwi, 256, 256);
    convert_b_frag<<<256, 256, 0, stream>>>(efwh, f_efwh, 256, 256);
    convert_b_frag<<<256, 256, 0, stream>>>(ebwh, f_ebwh, 256, 256);
    convert_b_frag<<<256, 256, 0, stream>>>(dfwh, f_dfwh, 256, 256);
    convert_b_frag<<<256, 256, 0, stream>>>(dbwh, f_dbwh, 256, 256);
    convert_b_frag<<<64000, 256, 0, stream>>>(dense_w, wdense, 512, 32000);

    // 2) Embedding gather + encoder input projections (both dirs).
    embed_proj<<<dim3(64, 2, 2), 256, 0, stream>>>(tokens, emb, f_efwi, f_ebwi,
                                                   projF, projB);
    // 3) Encoder scans -> ctx.
    enc_scan<<<2, 256, 0, stream>>>(projF, projB, f_efwh, f_ebwh, efb, ebb, ctx);
    // 4) Decoder constant input projection.
    dec_proj<<<64, 256, 0, stream>>>(ctx, dfwi, dfb, dbwi, dbb, pdf, pdb);
    // 5) Decoder scans -> bf16 A-matrix for Dense.
    dec_scan<<<2, 256, 0, stream>>>(pdf, pdb, f_dfwh, f_dbwh, decA);
    // 6) Dense logits (268 GFLOP, bf16 WMMA).
    dense_kernel<<<dim3(512, 125), 256, 0, stream>>>(decA, wdense, dense_b, out);
    // 7) Softmax in place.
    softmax_kernel<<<8192, 256, 0, stream>>>(out);
}
